// EnhancedSpatialGraphEncoder_50680614093456
// MI455X (gfx1250) — matrix operand order repaired and checked by reading, not compiled, not run on hardware
//
#include <hip/hip_runtime.h>
#include <hip/hip_bf16.h>

#define NN    10000   // nodes
#define EE    80000   // edges (w/o self loops)
#define DM    512     // d_model
#define HCW   1024    // H*C concat width
#define NEGS  0.2f

typedef __attribute__((ext_vector_type(16))) _Float16 v16h;
typedef __attribute__((ext_vector_type(8)))  float    v8f;
typedef unsigned int u32;

union FragU { v16h v; uint4 q[2]; };

__device__ __forceinline__ float lrelu(float v) { return v > 0.f ? v : NEGS * v; }

// ---------------- utility kernels ----------------

__global__ void f32_to_f16_k(const float* __restrict__ in, _Float16* __restrict__ out, int n) {
  for (int i = blockIdx.x * blockDim.x + threadIdx.x; i < n; i += gridDim.x * blockDim.x)
    out[i] = (_Float16)in[i];
}

// W [K, Nc] f32  ->  Wt [Nc, K] f16  (tiled transpose through LDS)
__global__ void transpose_w_k(const float* __restrict__ W, _Float16* __restrict__ Wt, int K, int Nc) {
  __shared__ float tile[32][33];
  int bk = blockIdx.x * 32, bn = blockIdx.y * 32;
  int tx = threadIdx.x, ty = threadIdx.y;            // 32 x 8
  for (int r = ty; r < 32; r += 8) {
    int k = bk + r, n = bn + tx;
    tile[r][tx] = (k < K && n < Nc) ? W[(size_t)k * Nc + n] : 0.f;
  }
  __syncthreads();
  for (int r = ty; r < 32; r += 8) {
    int n = bn + r, k = bk + tx;
    if (n < Nc && k < K) Wt[(size_t)n * K + k] = (_Float16)tile[tx][r];
  }
}

// ---------------- WMMA GEMM:  C[M,Nc](f32) = A[M,K](f16) * Wt[Nc,K](f16)^T -------------
// block: 256 thr = 8 waves, block tile 64x128, wave tile 32x32 (2x2 WMMA 16x16x32 f16)
__global__ __launch_bounds__(256) void gat_gemm_wmma_k(
    const _Float16* __restrict__ A, const _Float16* __restrict__ BT,
    float* __restrict__ C, int M, int K, int Nc) {
  int tid  = threadIdx.x;
  int wave = tid >> 5, lane = tid & 31;
  int wm = wave >> 2, wn = wave & 3;                 // 2 x 4 waves
  int m0 = blockIdx.x * 64 + wm * 32;
  int n0 = blockIdx.y * 128 + wn * 32;
  int lh = lane >> 4, lm = lane & 15;                // lane half / lane mod 16

  v8f acc[2][2] = {};
  const u32* A32 = (const u32*)A;
  const u32* B32 = (const u32*)BT;

  for (int k0 = 0; k0 < K; k0 += 32) {
    FragU fa[2], fb[2];
#pragma unroll
    for (int i = 0; i < 2; ++i) {
      // A fragment (16x32 f16): lane half lh -> K base lh*8; dwords 0-3 = K pairs 0..7,
      // dwords 4-7 = K pairs 16..23 (relative). Row = m0 + i*16 + lm.
      int row = m0 + i * 16 + lm; if (row >= M) row = 0;
      const u32* ap = A32 + (size_t)row * (K >> 1) + (k0 >> 1) + lh * 4;
      fa[i].q[0] = *(const uint4*)(ap);
      fa[i].q[1] = *(const uint4*)(ap + 8);
      // B fragment (32x16 f16): column n = lm, same K order as A: K base lh*8,
      // dwords 0-3 = K 0..7, dwords 4-7 = K 16..23 (relative), from BT[n][k].
      int col = n0 + i * 16 + lm;
      const u32* bp = B32 + (size_t)col * (K >> 1) + (k0 >> 1) + lh * 4;
      fb[i].q[0] = *(const uint4*)(bp);
      fb[i].q[1] = *(const uint4*)(bp + 8);
    }
#pragma unroll
    for (int i = 0; i < 2; ++i)
#pragma unroll
      for (int j = 0; j < 2; ++j)
        acc[i][j] = __builtin_amdgcn_wmma_f32_16x16x32_f16(
            false, fa[i].v, false, fb[j].v, (short)0, acc[i][j], false, false);
  }
  // C/D layout: lane l: col = lm, VGPR r -> row r + 8*lh
#pragma unroll
  for (int i = 0; i < 2; ++i)
#pragma unroll
    for (int j = 0; j < 2; ++j) {
      int col = n0 + j * 16 + lm;
#pragma unroll
      for (int r = 0; r < 8; ++r) {
        int row = m0 + i * 16 + r + 8 * lh;
        if (row < M) C[(size_t)row * Nc + col] = acc[i][j][r];
      }
    }
}

// ---------------- attention scores: a_s[i,h] = <h[i, h*256:...], a_src[h,:]> --------
__global__ __launch_bounds__(256) void attn_scores_k(
    const float* __restrict__ h, const float* __restrict__ a_src,
    const float* __restrict__ a_dst, float* __restrict__ as_, float* __restrict__ ad_) {
  int i = blockIdx.x, t = threadIdx.x;
  int hd = t >> 6;
  const float4 hv = *(const float4*)(h + (size_t)i * HCW + t * 4);
  const float4 sv = *(const float4*)(a_src + t * 4);
  const float4 dv = *(const float4*)(a_dst + t * 4);
  __shared__ float ss[256], sd[256];
  ss[t] = hv.x * sv.x + hv.y * sv.y + hv.z * sv.z + hv.w * sv.w;
  sd[t] = hv.x * dv.x + hv.y * dv.y + hv.z * dv.z + hv.w * dv.w;
  __syncthreads();
  for (int off = 32; off > 0; off >>= 1) {
    if ((t & 63) < off) { ss[t] += ss[t + off]; sd[t] += sd[t + off]; }
    __syncthreads();
  }
  if ((t & 63) == 0) { as_[i * 4 + hd] = ss[t]; ad_[i * 4 + hd] = sd[t]; }
}

// ---------------- CSR build ----------------
__global__ void deg_init_k(int* deg, int n) {
  int i = blockIdx.x * blockDim.x + threadIdx.x;
  if (i < n) deg[i] = 1;   // self loop
}
__global__ void deg_count_k(const int* __restrict__ dst, int e, int* deg) {
  int i = blockIdx.x * blockDim.x + threadIdx.x;
  if (i < e) atomicAdd(&deg[dst[i]], 1);
}
__global__ __launch_bounds__(1024) void scan_k(const int* __restrict__ deg, int* __restrict__ rowptr, int n) {
  __shared__ int s[1024];
  int t = threadIdx.x;
  int chunk = (n + 1023) >> 10;
  int b = t * chunk, e = b + chunk; if (e > n) e = n;
  int sum = 0;
  for (int i = b; i < e; ++i) sum += deg[i];
  s[t] = sum; __syncthreads();
  for (int off = 1; off < 1024; off <<= 1) {
    int v = (t >= off) ? s[t - off] : 0;
    __syncthreads();
    s[t] += v;
    __syncthreads();
  }
  int run = (t == 0) ? 0 : s[t - 1];
  for (int i = b; i < e; ++i) { rowptr[i] = run; run += deg[i]; }
  if (t == 1023) rowptr[n] = s[1023];
}
__global__ void cursor_init_k(const int* __restrict__ rowptr, int* cursor, int n) {
  int i = blockIdx.x * blockDim.x + threadIdx.x;
  if (i < n) cursor[i] = rowptr[i];
}
__global__ void fill_self_k(int* cursor, int* csr, int n) {
  int i = blockIdx.x * blockDim.x + threadIdx.x;
  if (i < n) { int p = atomicAdd(&cursor[i], 1); csr[p] = i; }
}
__global__ void fill_edges_k(const int* __restrict__ src, const int* __restrict__ dst,
                             int e, int* cursor, int* csr) {
  int i = blockIdx.x * blockDim.x + threadIdx.x;
  if (i < e) { int p = atomicAdd(&cursor[dst[i]], 1); csr[p] = src[i]; }
}

// ---------------- fused segment softmax + aggregation + bias + relu + f16 requant ----
__global__ __launch_bounds__(256) void gat_aggregate_k(
    const float* __restrict__ h, const float* __restrict__ as_, const float* __restrict__ ad_,
    const int* __restrict__ rowptr, const int* __restrict__ csr,
    const float* __restrict__ bias, _Float16* __restrict__ out16) {
  int i = blockIdx.x, t = threadIdx.x;
  int hd = t >> 6;                       // head for this thread's 4 columns
  int beg = rowptr[i], end = rowptr[i + 1];
  float adi = ad_[i * 4 + hd];
  __shared__ float sm[256];
  __shared__ float mh[4], dh[4];

  // pass 1: per-head max over incoming edges (64 threads per head)
  float lmax = -3.402823e38f;
  for (int j = beg + (t & 63); j < end; j += 64) {
    int s = csr[j];
    lmax = fmaxf(lmax, lrelu(as_[s * 4 + hd] + adi));
  }
  sm[t] = lmax; __syncthreads();
  for (int off = 32; off > 0; off >>= 1) {
    if ((t & 63) < off) sm[t] = fmaxf(sm[t], sm[t + off]);
    __syncthreads();
  }
  if ((t & 63) == 0) mh[hd] = sm[t];
  __syncthreads();
  float m = mh[hd];

  // pass 2: per-head softmax denominator
  float lden = 0.f;
  for (int j = beg + (t & 63); j < end; j += 64) {
    int s = csr[j];
    lden += __expf(lrelu(as_[s * 4 + hd] + adi) - m);
  }
  __syncthreads();
  sm[t] = lden; __syncthreads();
  for (int off = 32; off > 0; off >>= 1) {
    if ((t & 63) < off) sm[t] += sm[t + off];
    __syncthreads();
  }
  if ((t & 63) == 0) dh[hd] = sm[t];
  __syncthreads();
  float dn = dh[hd] + 1e-16f;

  // pass 3: weighted sum of source features; each thread owns 4 contiguous columns
  float a0 = 0.f, a1 = 0.f, a2 = 0.f, a3 = 0.f;
  for (int j = beg; j < end; ++j) {
    int s = csr[j];
    float alpha = __expf(lrelu(as_[s * 4 + hd] + adi) - m) / dn;
    const float4 hv = *(const float4*)(h + (size_t)s * HCW + t * 4);
    a0 += hv.x * alpha; a1 += hv.y * alpha; a2 += hv.z * alpha; a3 += hv.w * alpha;
  }
  int c = t * 4;
  const float4 bv = *(const float4*)(bias + c);
  float v0 = fmaxf(a0 + bv.x, 0.f), v1 = fmaxf(a1 + bv.y, 0.f);
  float v2 = fmaxf(a2 + bv.z, 0.f), v3 = fmaxf(a3 + bv.w, 0.f);
  _Float16* op = out16 + (size_t)i * HCW + c;
  op[0] = (_Float16)v0; op[1] = (_Float16)v1; op[2] = (_Float16)v2; op[3] = (_Float16)v3;
}

// ---------------- global mean pool (one block per column) ----------------
__global__ __launch_bounds__(256) void mean_pool_k(const _Float16* __restrict__ y, float* __restrict__ g) {
  int c = blockIdx.x, t = threadIdx.x;
  float s = 0.f;
  for (int r = t; r < NN; r += 256) s += (float)y[(size_t)r * HCW + c];
  __shared__ float sm[256];
  sm[t] = s; __syncthreads();
  for (int off = 128; off > 0; off >>= 1) {
    if (t < off) sm[t] += sm[t + off];
    __syncthreads();
  }
  if (t == 0) g[c] = sm[0] * (1.f / (float)NN);
}

// ---------------- head: y = g@Wo + bo, then layernorm ----------------
__global__ __launch_bounds__(512) void final_head_k(
    const float* __restrict__ g, const float* __restrict__ Wo, const float* __restrict__ bo,
    const float* __restrict__ gamma, const float* __restrict__ beta, float* __restrict__ out) {
  int t = threadIdx.x;   // 512
  float acc = bo[t];
  for (int k = 0; k < HCW; ++k) acc += g[k] * Wo[(size_t)k * DM + t];
  __shared__ float red[512];
  red[t] = acc; __syncthreads();
  for (int off = 256; off > 0; off >>= 1) {
    if (t < off) red[t] += red[t + off];
    __syncthreads();
  }
  float mu = red[0] * (1.f / (float)DM);
  __syncthreads();
  float d = acc - mu;
  red[t] = d * d; __syncthreads();
  for (int off = 256; off > 0; off >>= 1) {
    if (t < off) red[t] += red[t + off];
    __syncthreads();
  }
  float var = red[0] * (1.f / (float)DM);
  out[t] = d * rsqrtf(var + 1e-5f) * gamma[t] + beta[t];
}

// ---------------- host ----------------

extern "C" void kernel_launch(void* const* d_in, const int* in_sizes, int n_in,
                              void* d_out, int out_size, void* d_ws, size_t ws_size,
                              hipStream_t stream) {
  // ---- resolve input layout (insertion order vs jax sorted-key flatten) ----
  const float* x = nullptr; const int* ei = nullptr;
  const float *W[4], *asrc[4], *adst[4], *bvec[4];
  const float *Wo, *bo, *gamma, *beta;

  if (in_sizes[0] == 2 * EE) {
    // fully sorted leaves: edge_index, Wo, beta, bo, gamma, [W,a_dst,a_src,b]x4, x
    ei    = (const int*)d_in[0];
    Wo    = (const float*)d_in[1];
    beta  = (const float*)d_in[2];
    bo    = (const float*)d_in[3];
    gamma = (const float*)d_in[4];
    for (int l = 0; l < 4; ++l) {
      W[l]    = (const float*)d_in[5 + 4 * l];
      adst[l] = (const float*)d_in[6 + 4 * l];
      asrc[l] = (const float*)d_in[7 + 4 * l];
      bvec[l] = (const float*)d_in[8 + 4 * l];
    }
    x = (const float*)d_in[21];
  } else if (in_sizes[3] == DM) {
    // top insertion, params sorted: x, ei, Wo, beta, bo, gamma, [W,a_dst,a_src,b]x4
    x     = (const float*)d_in[0];
    ei    = (const int*)d_in[1];
    Wo    = (const float*)d_in[2];
    beta  = (const float*)d_in[3];
    bo    = (const float*)d_in[4];
    gamma = (const float*)d_in[5];
    for (int l = 0; l < 4; ++l) {
      W[l]    = (const float*)d_in[6 + 4 * l];
      adst[l] = (const float*)d_in[7 + 4 * l];
      asrc[l] = (const float*)d_in[8 + 4 * l];
      bvec[l] = (const float*)d_in[9 + 4 * l];
    }
  } else {
    // full insertion order: x, ei, [W,a_src,a_dst,b]x4, Wo, bo, gamma, beta
    x  = (const float*)d_in[0];
    ei = (const int*)d_in[1];
    for (int l = 0; l < 4; ++l) {
      W[l]    = (const float*)d_in[2 + 4 * l];
      asrc[l] = (const float*)d_in[3 + 4 * l];
      adst[l] = (const float*)d_in[4 + 4 * l];
      bvec[l] = (const float*)d_in[5 + 4 * l];
    }
    Wo    = (const float*)d_in[18];
    bo    = (const float*)d_in[19];
    gamma = (const float*)d_in[20];
    beta  = (const float*)d_in[21];
  }
  const int* esrc = ei;
  const int* edst = ei + EE;

  // ---- carve workspace ----
  char* p = (char*)d_ws;
  auto carve = [&](size_t bytes) -> void* {
    void* r = (void*)p;
    p += (bytes + 255) & ~(size_t)255;
    return r;
  };
  float*    h32    = (float*)carve((size_t)NN * HCW * 4);   // GEMM output (pre-aggregation features)
  _Float16* f16a   = (_Float16*)carve((size_t)NN * HCW * 2);
  _Float16* f16b   = (_Float16*)carve((size_t)NN * HCW * 2);
  _Float16* wt16   = (_Float16*)carve((size_t)HCW * HCW * 2);
  float*    as_    = (float*)carve((size_t)NN * 4 * 4);
  float*    ad_    = (float*)carve((size_t)NN * 4 * 4);
  int*      rowptr = (int*)carve((size_t)(NN + 1) * 4);
  int*      cursor = (int*)carve((size_t)NN * 4);           // doubles as deg[]
  int*      csr    = (int*)carve((size_t)(EE + NN) * 4);
  float*    g      = (float*)carve((size_t)HCW * 4);
  (void)ws_size; (void)n_in; (void)out_size;

  // ---- CSR by destination (deg stored in cursor buffer, then overwritten) ----
  deg_init_k<<<(NN + 255) / 256, 256, 0, stream>>>(cursor, NN);
  deg_count_k<<<(EE + 255) / 256, 256, 0, stream>>>(edst, EE, cursor);
  scan_k<<<1, 1024, 0, stream>>>(cursor, rowptr, NN);
  cursor_init_k<<<(NN + 255) / 256, 256, 0, stream>>>(rowptr, cursor, NN);
  fill_self_k<<<(NN + 255) / 256, 256, 0, stream>>>(cursor, csr, NN);
  fill_edges_k<<<(EE + 255) / 256, 256, 0, stream>>>(esrc, edst, EE, cursor, csr);

  // ---- layer 0 input to f16 ----
  f32_to_f16_k<<<2048, 256, 0, stream>>>(x, f16a, NN * DM);

  const int KL[4] = {DM, HCW, HCW, HCW};
  const _Float16* in16 = f16a;
  _Float16* out16 = f16b;
  for (int l = 0; l < 4; ++l) {
    int K = KL[l];
    transpose_w_k<<<dim3(K / 32, HCW / 32), dim3(32, 8), 0, stream>>>(W[l], wt16, K, HCW);
    gat_gemm_wmma_k<<<dim3((NN + 63) / 64, HCW / 128), 256, 0, stream>>>(in16, wt16, h32, NN, K, HCW);
    attn_scores_k<<<NN, 256, 0, stream>>>(h32, asrc[l], adst[l], as_, ad_);
    gat_aggregate_k<<<NN, 256, 0, stream>>>(h32, as_, ad_, rowptr, csr, bvec[l], out16);
    const _Float16* tmp = in16; in16 = out16; out16 = (_Float16*)tmp;
  }

  mean_pool_k<<<HCW, 256, 0, stream>>>(in16, g);
  final_head_k<<<1, 512, 0, stream>>>(g, Wo, bo, gamma, beta, (float*)d_out);
}
